// CausalSelfAttention_84868553769166
// MI455X (gfx1250) — compile-verified, matrix operand order checked by reading, not run on hardware
//
#include <hip/hip_runtime.h>

// ---------------- problem constants ----------------
#define B_  2
#define T_  2048
#define C_  2048
#define H_  16
#define G_  4
#define HS_ 128
#define QKV_DIM_ (((H_) + 2*(G_)) * (HS_))       // 3072
#define ODIM_    ((QKV_DIM_) + (H_))             // 3088
#define NTOK_    ((B_) * (T_))                   // 4096
#define EPS_     1e-5f
#define SCALE_   0.08838834764831845f            // 1/sqrt(128)

typedef __attribute__((ext_vector_type(16))) __bf16 bf16x16;
typedef __attribute__((ext_vector_type(8)))  __bf16 bf16x8;
typedef __attribute__((ext_vector_type(8)))  float  f32x8;
typedef __attribute__((ext_vector_type(4)))  unsigned int u32x4;
typedef __attribute__((ext_vector_type(8)))  unsigned int u32x8;

// fp32 -> bf16 (round to nearest even), raw bits
__device__ __forceinline__ unsigned short f2bf(float f) {
  unsigned int u = __float_as_uint(f);
  u += 0x7fffu + ((u >> 16) & 1u);
  return (unsigned short)(u >> 16);
}

__device__ __forceinline__ f32x8 wmma_bf16(bf16x16 a, bf16x16 b, f32x8 c) {
  return __builtin_amdgcn_wmma_f32_16x16x32_bf16(
      false, a, false, b, (short)0, c, false, false);
}

// Load one 16x32 bf16 fragment (row/col = lane&15; K chunks
// {0..7,16..23} for lanes 0-15, {8..15,24..31} for lanes 16-31).
__device__ __forceinline__ bf16x16 load_frag(const unsigned short* base, int ld, int lane) {
  const int half = (lane >> 4) & 1;
  const int idx  = lane & 15;
  const unsigned short* r = base + (size_t)idx * (size_t)ld;
  bf16x8 lo = *(const bf16x8*)(r + half * 8);
  bf16x8 hi = *(const bf16x8*)(r + 16 + half * 8);
  return __builtin_shufflevector(lo, hi, 0,1,2,3,4,5,6,7,8,9,10,11,12,13,14,15);
}

// ---- Tensor Data Mover: 2D tile (2-byte elements) global -> LDS ----
// D# per CDNA5 ISA ch.8; tile exactly covers region (tensor_dim == tile_dim).
__device__ __forceinline__ void tdm_load_2d(unsigned lds_off,
                                            const unsigned short* gptr,
                                            unsigned tile_w, unsigned tile_h,
                                            unsigned stride_elems) {
  unsigned long long ga = (unsigned long long)gptr;
  u32x4 g0;
  g0[0] = 0x1u;                                        // count=1, user desc
  g0[1] = lds_off;                                     // LDS byte address
  g0[2] = (unsigned)(ga & 0xffffffffu);                // global_addr[31:0]
  g0[3] = (unsigned)((ga >> 32) & 0x1ffffffu)          // global_addr[56:32]
          | 0x80000000u;                               // type=2 ("image")
  u32x8 g1;
  g1[0] = 0x00010000u;                                 // mask=0, data_size=1 (2B)
  g1[1] = (tile_w & 0xffffu) << 16;                    // tensor_dim0[15:0]
  g1[2] = ((tile_w >> 16) & 0xffffu)
          | ((tile_h & 0xffffu) << 16);                // dim0 hi | tensor_dim1 lo
  g1[3] = ((tile_h >> 16) & 0xffffu)
          | ((tile_w & 0xffffu) << 16);                // dim1 hi | tile_dim0
  g1[4] = (tile_h & 0xffffu);                          // tile_dim1 ; tile_dim2=0
  g1[5] = stride_elems;                                // tensor_dim0_stride[31:0]
  g1[6] = 0u;
  g1[7] = 0u;
  u32x4 gz = {0u, 0u, 0u, 0u};                         // groups 2/3 unused (2D)
  asm volatile("tensor_load_to_lds %0, %1, %2, %3"
               :: "s"(g0), "s"(g1), "s"(gz), "s"(gz)
               : "memory");
}

__device__ __forceinline__ unsigned lds_addr_of(const void* p) {
  return (unsigned)(unsigned long long)p;              // LDS aperture: addr[31:0]
}

// ---------------- kernel 1: fp32 -> bf16 convert ----------------
__global__ void cvt_bf16_kernel(const float* __restrict__ in,
                                unsigned short* __restrict__ out, int n) {
  int i = blockIdx.x * blockDim.x + threadIdx.x;
  if (i < n) out[i] = f2bf(in[i]);
}

// ------- kernels 2/5: cooperative TDM-staged bf16 GEMM, f32 out -------
// C[m][n] = sum_k A[m][k] * Bm[n][k]; block = 8 waves -> 128x128 tile.
// Wave 0 double-buffers 128x32 A/B slabs into LDS via TDM.
__global__ __launch_bounds__(256)
void gemm_bf16_tdm_kernel(const unsigned short* __restrict__ A,
                          const unsigned short* __restrict__ Bm,
                          float* __restrict__ Cout,
                          int M, int N, int K, int nblk_n) {
  __shared__ unsigned short sa[2][128 * 32];
  __shared__ unsigned short sb[2][128 * 32];

  const int lane = threadIdx.x & 31;
  const int wid  = threadIdx.x >> 5;
  const int mb = blockIdx.x / nblk_n;
  const int nb = blockIdx.x - mb * nblk_n;
  if (mb * 128 >= M) return;
  const int m0 = mb * 128;
  const int n0 = nb * 128;
  const int half = lane >> 4;
  const int col  = lane & 15;

  const unsigned short* Ab = A  + (size_t)m0 * (size_t)K;
  const unsigned short* Bb = Bm + (size_t)n0 * (size_t)K;
  const unsigned ldsa[2] = { lds_addr_of(&sa[0][0]), lds_addr_of(&sa[1][0]) };
  const unsigned ldsb[2] = { lds_addr_of(&sb[0][0]), lds_addr_of(&sb[1][0]) };

  f32x8 acc[8];
  #pragma unroll
  for (int j = 0; j < 8; ++j) acc[j] = (f32x8){};

  const int niter = K / 32;
  if (wid == 0) {                                  // prologue: tile 0
    tdm_load_2d(ldsa[0], Ab, 32u, 128u, (unsigned)K);
    tdm_load_2d(ldsb[0], Bb, 32u, 128u, (unsigned)K);
  }

  for (int i = 0; i < niter; ++i) {
    if (wid == 0) {
      if (i + 1 < niter) {                         // overlap DMA with compute
        int k1 = (i + 1) * 32;
        tdm_load_2d(ldsa[(i + 1) & 1], Ab + k1, 32u, 128u, (unsigned)K);
        tdm_load_2d(ldsb[(i + 1) & 1], Bb + k1, 32u, 128u, (unsigned)K);
        __builtin_amdgcn_s_wait_tensorcnt(2);      // oldest 2 (this tile) done
      } else {
        __builtin_amdgcn_s_wait_tensorcnt(0);
      }
    }
    __syncthreads();

    const unsigned short* pa = &sa[i & 1][0];
    const unsigned short* pb = &sb[i & 1][0];
    bf16x16 a = load_frag(pa + wid * 16 * 32, 32, lane);
    #pragma unroll
    for (int j = 0; j < 8; ++j) {
      bf16x16 b = load_frag(pb + j * 16 * 32, 32, lane);
      acc[j] = wmma_bf16(a, b, acc[j]);
    }
    __syncthreads();                               // reads done before overwrite
  }

  #pragma unroll
  for (int j = 0; j < 8; ++j) {
    int n = n0 + j * 16 + col;
    #pragma unroll
    for (int r = 0; r < 8; ++r) {
      int row = m0 + wid * 16 + r + half * 8;
      Cout[(size_t)row * N + n] = acc[j][r];
    }
  }
}

// -------- small direct-global GEMM for the ragged gate columns --------
template <int JN>
__global__ __launch_bounds__(256)
void gemm_bf16_kernel(const unsigned short* __restrict__ A,
                      const unsigned short* __restrict__ Bm,
                      float* __restrict__ Cout,
                      int M, int N, int K, int ntiles_n, int n_base) {
  const int lane = threadIdx.x & 31;
  const int wid  = threadIdx.x >> 5;
  const int w    = blockIdx.x * 8 + wid;
  const int mt = w / ntiles_n;
  const int nt = w - mt * ntiles_n;
  if (mt * 16 >= M) return;
  const int m0 = mt * 16;
  const int n0 = n_base + nt * (JN * 16);
  const int half = lane >> 4;
  const int col  = lane & 15;

  f32x8 acc[JN];
  #pragma unroll
  for (int j = 0; j < JN; ++j) acc[j] = (f32x8){};
  const unsigned short* Abase = A + (size_t)m0 * (size_t)K;

  for (int k0 = 0; k0 < K; k0 += 32) {
    if (k0 + 128 < K)
      __builtin_prefetch(Abase + (size_t)(lane & 15) * K + k0 + 128, 0, 3);
    bf16x16 a = load_frag(Abase + k0, K, lane);
    #pragma unroll
    for (int j = 0; j < JN; ++j) {
      bf16x16 b = load_frag(Bm + (size_t)(n0 + j * 16) * K + k0, K, lane);
      acc[j] = wmma_bf16(a, b, acc[j]);
    }
  }
  #pragma unroll
  for (int j = 0; j < JN; ++j) {
    int n = n0 + j * 16 + col;
    #pragma unroll
    for (int r = 0; r < 8; ++r) {
      int row = m0 + r + half * 8;
      Cout[(size_t)row * N + n] = acc[j][r];
    }
  }
}

// ---------------- kernel 3: RMSnorm + RoPE + layout ----------------
__global__ __launch_bounds__(256)
void postproc_kernel(const float* __restrict__ qkv,
                     const float* __restrict__ cosb, const float* __restrict__ sinb,
                     const float* __restrict__ qw, const float* __restrict__ kw,
                     unsigned short* __restrict__ qb,   // [B,H,T,HS]
                     unsigned short* __restrict__ kb,   // [B,G,T,HS]
                     unsigned short* __restrict__ vtb)  // [B,G,HS,T]
{
  const int lane = threadIdx.x & 31;
  const int wid  = threadIdx.x >> 5;
  const int w    = blockIdx.x * 8 + wid;
  const int tok  = w / 24;
  const int unit = w - tok * 24;
  if (tok >= NTOK_) return;
  const int b = tok / T_;
  const int t = tok - b * T_;

  if (unit < H_ + G_) {
    const int isq  = unit < H_ ? 1 : 0;
    const int head = isq ? unit : (unit - H_);
    const float* src = qkv + (size_t)tok * ODIM_ + (isq ? head * HS_ : (H_ * HS_ + head * HS_));
    const float* wgt = isq ? qw : kw;
    float v0 = src[lane], v1 = src[lane + 32], v2 = src[lane + 64], v3 = src[lane + 96];
    float ss = v0*v0 + v1*v1 + v2*v2 + v3*v3;
    #pragma unroll
    for (int m = 1; m < 32; m <<= 1) ss += __shfl_xor(ss, m, 32);
    float rinv = rsqrtf(ss * (1.0f / HS_) + EPS_);
    v0 *= rinv * wgt[lane];
    v1 *= rinv * wgt[lane + 32];
    v2 *= rinv * wgt[lane + 64];
    v3 *= rinv * wgt[lane + 96];
    float c0 = cosb[t * 64 + lane],      s0 = sinb[t * 64 + lane];
    float c1 = cosb[t * 64 + lane + 32], s1 = sinb[t * 64 + lane + 32];
    float o0 = v0 * c0 - v2 * s0;
    float o2 = v2 * c0 + v0 * s0;
    float o1 = v1 * c1 - v3 * s1;
    float o3 = v3 * c1 + v1 * s1;
    unsigned short* dst = isq
      ? qb + ((size_t)(b * H_ + head) * T_ + t) * HS_
      : kb + ((size_t)(b * G_ + head) * T_ + t) * HS_;
    dst[lane]      = f2bf(o0);
    dst[lane + 32] = f2bf(o1);
    dst[lane + 64] = f2bf(o2);
    dst[lane + 96] = f2bf(o3);
  } else {
    const int g = unit - H_ - G_;
    const float* src = qkv + (size_t)tok * ODIM_ + (H_ + G_) * HS_ + g * HS_;
    unsigned short* dst = vtb + (size_t)(b * G_ + g) * HS_ * T_;
    #pragma unroll
    for (int e = 0; e < 4; ++e) {
      int hs = lane + e * 32;
      dst[(size_t)hs * T_ + t] = f2bf(src[hs]);
    }
  }
}

// ------- kernel 4: flash attention (double-buffered TDM K/V) -------
// block = 8 waves sharing (b,h) and 128 q rows; wave owns 16 rows.
__global__ __launch_bounds__(256)
void attn_kernel(const unsigned short* __restrict__ qb,
                 const unsigned short* __restrict__ kb,
                 const unsigned short* __restrict__ vtb,
                 const float* __restrict__ qkv,           // gate
                 unsigned short* __restrict__ aob)        // [NTOK, H*HS]
{
  __shared__ unsigned short sk[2][32 * 128];   // K tile: rows s-local, ld=128
  __shared__ unsigned short sv[2][128 * 32];   // V^T tile: rows hs, ld=32
  __shared__ unsigned short sp[8][16 * 32];    // per-wave P staging

  const int lane = threadIdx.x & 31;
  const int wid  = threadIdx.x >> 5;
  const int tgrp = blockIdx.x & 15;
  const int h    = (blockIdx.x >> 4) & 15;
  const int b    = blockIdx.x >> 8;
  const int g    = h >> 2;
  const int t0   = tgrp * 128 + wid * 16;
  const int half = lane >> 4;
  const int col  = lane & 15;

  const unsigned short* qhead = qb  + (size_t)(b * H_ + h) * T_ * HS_;
  const unsigned short* khead = kb  + (size_t)(b * G_ + g) * T_ * HS_;
  const unsigned short* vhead = vtb + (size_t)(b * G_ + g) * HS_ * T_;
  const unsigned ldsk[2] = { lds_addr_of(&sk[0][0]), lds_addr_of(&sk[1][0]) };
  const unsigned ldsv[2] = { lds_addr_of(&sv[0][0]), lds_addr_of(&sv[1][0]) };

  bf16x16 qf[4];
  #pragma unroll
  for (int kk = 0; kk < 4; ++kk)
    qf[kk] = load_frag(qhead + (size_t)t0 * HS_ + kk * 32, HS_, lane);

  f32x8 oacc[8];
  #pragma unroll
  for (int j = 0; j < 8; ++j) oacc[j] = (f32x8){};
  float mrow[8], lrow[8];
  #pragma unroll
  for (int r = 0; r < 8; ++r) { mrow[r] = -1e30f; lrow[r] = 0.0f; }

  const int my_smax = t0 + 16;
  const int niter   = tgrp * 4 + 4;              // (tgrp*128+128)/32, uniform

  if (wid == 0) {                                // prologue: s-block 0
    tdm_load_2d(ldsk[0], khead, 128u, 32u, (unsigned)HS_);
    tdm_load_2d(ldsv[0], vhead, 32u, 128u, (unsigned)T_);
  }

  for (int i = 0; i < niter; ++i) {
    const int s0 = i * 32;
    if (wid == 0) {
      if (i + 1 < niter) {
        int s1b = s0 + 32;
        tdm_load_2d(ldsk[(i + 1) & 1], khead + (size_t)s1b * HS_, 128u, 32u, (unsigned)HS_);
        tdm_load_2d(ldsv[(i + 1) & 1], vhead + s1b, 32u, 128u, (unsigned)T_);
        __builtin_amdgcn_s_wait_tensorcnt(2);
      } else {
        __builtin_amdgcn_s_wait_tensorcnt(0);
      }
    }
    __syncthreads();

    if (s0 < my_smax) {
      const unsigned short* pk = &sk[i & 1][0];
      const unsigned short* pv = &sv[i & 1][0];
      // ---- S = Q K^T (16 x 32) ----
      f32x8 sc0 = {}, sc1 = {};
      #pragma unroll
      for (int kk = 0; kk < 4; ++kk) {
        bf16x16 b0 = load_frag(pk + kk * 32, 128, lane);
        sc0 = wmma_bf16(qf[kk], b0, sc0);
        bf16x16 b1 = load_frag(pk + 16 * 128 + kk * 32, 128, lane);
        sc1 = wmma_bf16(qf[kk], b1, sc1);
      }
      #pragma unroll
      for (int r = 0; r < 8; ++r) { sc0[r] *= SCALE_; sc1[r] *= SCALE_; }
      if (s0 + 31 >= t0) {
        #pragma unroll
        for (int r = 0; r < 8; ++r) {
          int trow = t0 + r + half * 8;
          if (s0 + col      > trow) sc0[r] = -1e30f;
          if (s0 + 16 + col > trow) sc1[r] = -1e30f;
        }
      }
      // ---- online softmax ----
      #pragma unroll
      for (int r = 0; r < 8; ++r) {
        float mb = fmaxf(sc0[r], sc1[r]);
        mb = fmaxf(mb, __shfl_xor(mb, 1, 16));
        mb = fmaxf(mb, __shfl_xor(mb, 2, 16));
        mb = fmaxf(mb, __shfl_xor(mb, 4, 16));
        mb = fmaxf(mb, __shfl_xor(mb, 8, 16));
        float mnew  = fmaxf(mrow[r], mb);
        float alpha = __expf(mrow[r] - mnew);
        float p0 = __expf(sc0[r] - mnew);
        float p1 = __expf(sc1[r] - mnew);
        float ps = p0 + p1;
        ps += __shfl_xor(ps, 1, 16);
        ps += __shfl_xor(ps, 2, 16);
        ps += __shfl_xor(ps, 4, 16);
        ps += __shfl_xor(ps, 8, 16);
        lrow[r] = lrow[r] * alpha + ps;
        mrow[r] = mnew;
        int row = r + half * 8;
        sp[wid][row * 32 + col]      = f2bf(p0);
        sp[wid][row * 32 + 16 + col] = f2bf(p1);
        #pragma unroll
        for (int j = 0; j < 8; ++j) oacc[j][r] *= alpha;
      }
      asm volatile("s_wait_dscnt 0" ::: "memory");
      bf16x16 pf = load_frag(&sp[wid][0], 32, lane);
      // ---- O += P V (16 x 128) ----
      #pragma unroll
      for (int j = 0; j < 8; ++j) {
        bf16x16 vf = load_frag(pv + (j * 16) * 32, 32, lane);
        oacc[j] = wmma_bf16(pf, vf, oacc[j]);
      }
    }
    __syncthreads();
  }

  // ---- epilogue: 1/l, sigmoid(gate), store bf16 ----
  #pragma unroll
  for (int r = 0; r < 8; ++r) {
    int t = t0 + r + half * 8;
    float gate = qkv[(size_t)(b * T_ + t) * ODIM_ + QKV_DIM_ + h];
    float gs = 1.0f / (1.0f + __expf(-gate));
    float sc = gs / lrow[r];
    #pragma unroll
    for (int j = 0; j < 8; ++j) {
      float v = oacc[j][r] * sc;
      aob[(size_t)(b * T_ + t) * (H_ * HS_) + h * HS_ + j * 16 + col] = f2bf(v);
    }
  }
}

// ---------------- host side ----------------
extern "C" void kernel_launch(void* const* d_in, const int* in_sizes, int n_in,
                              void* d_out, int out_size, void* d_ws, size_t ws_size,
                              hipStream_t stream) {
  const float* x      = (const float*)d_in[0];
  const float* cosb   = (const float*)d_in[1];
  const float* sinb   = (const float*)d_in[2];
  const float* W_attn = (const float*)d_in[3];
  const float* qw     = (const float*)d_in[4];
  const float* kw     = (const float*)d_in[5];
  const float* W_proj = (const float*)d_in[6];
  float* y = (float*)d_out;

  char* ws = (char*)d_ws;
  size_t off = 0;
  auto alloc = [&](size_t bytes) -> void* {
    off = (off + 255) & ~(size_t)255;
    void* p = ws + off;
    off += bytes;
    return p;
  };

  unsigned short* xb   = (unsigned short*)alloc((size_t)NTOK_ * C_ * 2);
  unsigned short* Wab  = (unsigned short*)alloc((size_t)ODIM_ * C_ * 2);
  unsigned short* Wpb  = (unsigned short*)alloc((size_t)C_ * (H_ * HS_) * 2);
  float*          qkvf = (float*)         alloc((size_t)NTOK_ * ODIM_ * 4);
  unsigned short* qb   = (unsigned short*)alloc((size_t)NTOK_ * H_ * HS_ * 2);
  unsigned short* kb   = (unsigned short*)alloc((size_t)NTOK_ * G_ * HS_ * 2);
  unsigned short* vtb  = (unsigned short*)alloc((size_t)NTOK_ * G_ * HS_ * 2);
  unsigned short* aob  = (unsigned short*)alloc((size_t)NTOK_ * H_ * HS_ * 2);
  (void)ws_size;

  // 1) convert to bf16
  {
    int n1 = NTOK_ * C_;
    cvt_bf16_kernel<<<(n1 + 255) / 256, 256, 0, stream>>>(x, xb, n1);
    int n2 = ODIM_ * C_;
    cvt_bf16_kernel<<<(n2 + 255) / 256, 256, 0, stream>>>(W_attn, Wab, n2);
    int n3 = C_ * H_ * HS_;
    cvt_bf16_kernel<<<(n3 + 255) / 256, 256, 0, stream>>>(W_proj, Wpb, n3);
  }

  // 2) QKV GEMM: TDM-staged 128x128 blocks + 16-wide gate remainder
  {
    int nbn = QKV_DIM_ / 128;                    // 24
    int blocks = (NTOK_ / 128) * nbn;            // 32*24 = 768
    gemm_bf16_tdm_kernel<<<blocks, 256, 0, stream>>>(
        xb, Wab, qkvf, NTOK_, ODIM_, C_, nbn);
    int tiles1 = NTOK_ / 16;                     // gate columns 3072..3087
    gemm_bf16_kernel<1><<<(tiles1 + 7) / 8, 256, 0, stream>>>(
        xb, Wab, qkvf, NTOK_, ODIM_, C_, 1, QKV_DIM_);
  }

  // 3) RMSnorm + RoPE + layout
  {
    int waves = NTOK_ * 24;
    postproc_kernel<<<waves / 8, 256, 0, stream>>>(
        qkvf, cosb, sinb, qw, kw, qb, kb, vtb);
  }

  // 4) flash attention (double-buffered TDM staging)
  {
    int blocks = B_ * H_ * (T_ / 128);           // 512
    attn_kernel<<<blocks, 256, 0, stream>>>(qb, kb, vtb, qkvf, aob);
  }

  // 5) output projection: y = aob @ W_proj^T (TDM-staged)
  {
    int nbn = C_ / 128;                          // 16
    int blocks = (NTOK_ / 128) * nbn;            // 512
    gemm_bf16_tdm_kernel<<<blocks, 256, 0, stream>>>(
        aob, Wpb, y, NTOK_, C_, C_, nbn);
  }
}
